// CausalAttentionHead_63264868270183
// MI455X (gfx1250) — matrix operand order repaired and checked by reading, not compiled, not run on hardware
//
#include <hip/hip_runtime.h>

typedef __attribute__((ext_vector_type(16))) _Float16 v16h;
typedef __attribute__((ext_vector_type(8)))  _Float16 v8h;
typedef __attribute__((ext_vector_type(8)))  float    v8f;
typedef __attribute__((ext_vector_type(4)))  float    v4f;

#define BATCH 8
#define SLEN  2048
#define EDIM  768
#define HDIM  128
#define WSZ   (HDIM * EDIM)   // 98304 elements per weight matrix

static __device__ __forceinline__ v8f wmma_f16(v16h a, v16h b, v8f c) {
  // D = A(16x32 f16) * B(32x16 f16) + C(16x16 f32)
  return __builtin_amdgcn_wmma_f32_16x16x32_f16(false, a, false, b, (short)0, c,
                                                false, false);
}

// ---- A-operand fragment (16x32, f16): lane holds row m = lane%16.
// element i -> k = (lane/16)*16 + i.
// p must already point at row_base + (lane>>4)*16 elements.
static __device__ __forceinline__ v16h fragA_h(const _Float16* p) {
  v8h a = *(const v8h*)(p);
  v8h b = *(const v8h*)(p + 8);
  v16h f;
#pragma unroll
  for (int i = 0; i < 8; ++i) { f[i] = a[i]; f[8 + i] = b[i]; }
  return f;
}

static __device__ __forceinline__ v16h fragA_f32(const float* p, float s) {
  v16h f;
#pragma unroll
  for (int i = 0; i < 8; ++i) f[i] = (_Float16)(p[i] * s);
#pragma unroll
  for (int i = 0; i < 8; ++i) f[8 + i] = (_Float16)(p[8 + i] * s);
  return f;
}

// ---- B-operand fragment (32x16, f16): lane holds column n = lane%16.
// element i -> k = (lane/16)*16 + i.  p points at col_base + (lane>>4)*16.
static __device__ __forceinline__ v16h fragB_h(const _Float16* p) {
  v8h a = *(const v8h*)(p);
  v8h b = *(const v8h*)(p + 8);
  v16h f;
#pragma unroll
  for (int i = 0; i < 8; ++i) { f[i] = a[i]; f[8 + i] = b[i]; }
  return f;
}

// ====================================================================
// Stage 0: one-shot f32->f16 conversion of the three weight matrices.
// Wh layout: [3][128][768] f16.  4 elements per thread.
// ====================================================================
__global__ __launch_bounds__(256) void wconv_kernel(
    const float* __restrict__ Wq, const float* __restrict__ Wk,
    const float* __restrict__ Wv, _Float16* __restrict__ Wh)
{
  const int i4 = (blockIdx.x * 256 + threadIdx.x) * 4;   // element index
  const int which = i4 / WSZ;
  const int off   = i4 % WSZ;
  const float* src = (which == 0) ? Wq : (which == 1) ? Wk : Wv;
  const v4f v = *(const v4f*)(src + off);
#pragma unroll
  for (int i = 0; i < 4; ++i) Wh[i4 + i] = (_Float16)v[i];
}

// ====================================================================
// Stage 1: Q/K/V projections.  grid = (M/128, 3), block = 256 (8 waves).
// Wave w computes a 16x128 output tile with a 24-step K loop over E=768.
// which = blockIdx.y: 0->Q (scaled by 1/sqrt(H)), 1->K, 2->V (transposed).
// All 8 B-fragments of a K-step are loaded before the WMMA burst so the
// compiler can overlap loads with matrix ops (partial loadcnt waits).
// ====================================================================
__global__ __launch_bounds__(256) void qkv_kernel(
    const float* __restrict__ X, const float* __restrict__ pmask,
    const _Float16* __restrict__ Wh,
    _Float16* __restrict__ Qh, _Float16* __restrict__ Kh,
    _Float16* __restrict__ Vt)
{
  const int lane = threadIdx.x & 31;
  const int wave = threadIdx.x >> 5;
  const int lid  = lane & 15;   // row/col index within a 16-tile
  const int lh   = lane >> 4;   // lane-half
  const int which = blockIdx.y;
  const _Float16* W = Wh + which * WSZ;

  const int row0 = blockIdx.x * 128 + wave * 16;
  const int arow = row0 + lid;
  const float scale = pmask[arow];            // X masked -> affects q,k,v

  v8f acc[8] = {};

  const float* xp = X + arow * EDIM + lh * 16;
  const _Float16* wp = W + lid * EDIM + lh * 16;
#pragma unroll 1
  for (int k0 = 0; k0 < EDIM; k0 += 32) {
    v16h bf[8];
#pragma unroll
    for (int t = 0; t < 8; ++t)
      bf[t] = fragB_h(wp + t * (16 * EDIM) + k0);
    const v16h a = fragA_f32(xp + k0, scale);
#pragma unroll
    for (int t = 0; t < 8; ++t)
      acc[t] = wmma_f16(a, bf[t], acc[t]);
  }

  // C layout: VGPR r holds M = r + 8*(lane/16), lane%16 = N
  if (which < 2) {
    const float osc = (which == 0) ? 0.088388347648318447f : 1.0f;  // 1/sqrt(128)
    _Float16* dst = (which == 0) ? Qh : Kh;
#pragma unroll
    for (int t = 0; t < 8; ++t)
#pragma unroll
      for (int r = 0; r < 8; ++r) {
        const int mm = row0 + r + 8 * lh;
        dst[mm * HDIM + t * 16 + lid] = (_Float16)(acc[t][r] * osc);
      }
  } else {
    const int b  = row0 >> 11;
    const int sb = row0 & (SLEN - 1);
#pragma unroll
    for (int t = 0; t < 8; ++t)
#pragma unroll
      for (int r = 0; r < 8; ++r) {
        const int ss = sb + r + 8 * lh;
        Vt[b * (HDIM * SLEN) + (t * 16 + lid) * SLEN + ss] = (_Float16)acc[t][r];
      }
  }
}

// ====================================================================
// Stage 2: causal flash attention.  grid = B*(S/128), block = 256.
// Each wave owns 16 query rows; O accumulator 16x128 f32 in registers.
// Key loop in blocks of 32 up to the wave's causal frontier.
// K / V fragments are batch-loaded before each WMMA burst.
// ====================================================================
__global__ __launch_bounds__(256) void flash_attn_kernel(
    const _Float16* __restrict__ Qh, const _Float16* __restrict__ Kh,
    const _Float16* __restrict__ Vt, float* __restrict__ Out)
{
  __shared__ __align__(16) _Float16 Ptile[8 * 16 * 32];  // per-wave P staging
  const int lane = threadIdx.x & 31;
  const int wave = threadIdx.x >> 5;
  const int lid  = lane & 15;
  const int lh   = lane >> 4;
  const int b    = blockIdx.x >> 4;
  const int qblk = blockIdx.x & 15;
  const int qb   = qblk * 128 + wave * 16;

  const _Float16* Qb = Qh + b * (SLEN * HDIM);
  const _Float16* Kb = Kh + b * (SLEN * HDIM);
  const _Float16* Vb = Vt + b * (HDIM * SLEN);
  _Float16* Pl = Ptile + wave * (16 * 32);

  // Q fragments for the whole head dim (4 x K=32), kept in registers
  v16h qf[4];
#pragma unroll
  for (int kk = 0; kk < 4; ++kk)
    qf[kk] = fragA_h(Qb + (qb + lid) * HDIM + kk * 32 + lh * 16);

  v8f o[8] = {};
  float rmax[8], rsum[8];
#pragma unroll
  for (int r = 0; r < 8; ++r) { rmax[r] = -3.0e38f; rsum[r] = 0.0f; }

  const int q_hi = qb + 15;
#pragma unroll 1
  for (int j = 0; j <= q_hi; j += 32) {
    // ---- S tile = Q(16x128) x K_j^T(128x32), f32 accum.
    // Load all 8 K fragments first, then burst the WMMAs.
    v8f s0 = {}, s1 = {};
    {
      v16h kf[8];
#pragma unroll
      for (int kk = 0; kk < 4; ++kk) {
        kf[2 * kk]     = fragB_h(Kb + (j + lid) * HDIM + kk * 32 + lh * 16);
        kf[2 * kk + 1] = fragB_h(Kb + (j + 16 + lid) * HDIM + kk * 32 + lh * 16);
      }
#pragma unroll
      for (int kk = 0; kk < 4; ++kk) {
        s0 = wmma_f16(qf[kk], kf[2 * kk], s0);
        s1 = wmma_f16(qf[kk], kf[2 * kk + 1], s1);
      }
    }

    // ---- causal mask + online-softmax row max (rows live per-VGPR,
    // cols across the 16-lane half-groups -> xor-shuffle reduction)
    float nmax[8], alpha[8];
#pragma unroll
    for (int r = 0; r < 8; ++r) {
      const int mg = qb + r + 8 * lh;
      if (j + lid > mg)      s0[r] = -3.0e38f;
      if (j + 16 + lid > mg) s1[r] = -3.0e38f;
      float v = fmaxf(s0[r], s1[r]);
      v = fmaxf(v, __shfl_xor(v, 1, 32));
      v = fmaxf(v, __shfl_xor(v, 2, 32));
      v = fmaxf(v, __shfl_xor(v, 4, 32));
      v = fmaxf(v, __shfl_xor(v, 8, 32));
      const float nm = fmaxf(rmax[r], v);
      nmax[r]  = nm;
      alpha[r] = __expf(rmax[r] - nm);
      rmax[r]  = nm;
    }

    // ---- P = exp(S - nm): stage f16 P in LDS (C-layout -> A-layout),
    // accumulate row sums
#pragma unroll
    for (int r = 0; r < 8; ++r) {
      const float p0 = __expf(s0[r] - nmax[r]);
      const float p1 = __expf(s1[r] - nmax[r]);
      const int ml = r + 8 * lh;
      Pl[ml * 32 + lid]      = (_Float16)p0;
      Pl[ml * 32 + 16 + lid] = (_Float16)p1;
      float ps = p0 + p1;
      ps += __shfl_xor(ps, 1, 32);
      ps += __shfl_xor(ps, 2, 32);
      ps += __shfl_xor(ps, 4, 32);
      ps += __shfl_xor(ps, 8, 32);
      rsum[r] = rsum[r] * alpha[r] + ps;
    }

    // ---- rescale O by alpha (per row)
#pragma unroll
    for (int t = 0; t < 8; ++t)
#pragma unroll
      for (int r = 0; r < 8; ++r) o[t][r] *= alpha[r];

    // ---- PV: batch-load V fragments, read P back in A-layout, burst WMMAs.
    {
      v16h vf[8];
#pragma unroll
      for (int t = 0; t < 8; ++t)
        vf[t] = fragB_h(Vb + (t * 16 + lid) * SLEN + j + lh * 16);
      // same-wave LDS ops are in-order; fence compiler + data return
      asm volatile("s_wait_dscnt 0" ::: "memory");
      const v16h pf = fragA_h(Pl + lid * 32 + lh * 16);
#pragma unroll
      for (int t = 0; t < 8; ++t)
        o[t] = wmma_f16(pf, vf[t], o[t]);
    }
    asm volatile("" ::: "memory");  // keep next iter's P stores after loads
  }

  float* Ob = Out + b * (SLEN * HDIM);
#pragma unroll
  for (int t = 0; t < 8; ++t)
#pragma unroll
    for (int r = 0; r < 8; ++r) {
      const int mg = qb + r + 8 * lh;
      Ob[mg * HDIM + t * 16 + lid] = o[t][r] / rsum[r];
    }
}

// ====================================================================
extern "C" void kernel_launch(void* const* d_in, const int* in_sizes, int n_in,
                              void* d_out, int out_size, void* d_ws, size_t ws_size,
                              hipStream_t stream) {
  const float* X     = (const float*)d_in[0];
  const float* pmask = (const float*)d_in[1];
  const float* Wq    = (const float*)d_in[2];
  const float* Wk    = (const float*)d_in[3];
  const float* Wv    = (const float*)d_in[4];
  float* out = (float*)d_out;

  _Float16* Qh = (_Float16*)d_ws;                      // [B*S,128] f16 (pre-scaled)
  _Float16* Kh = Qh + (size_t)BATCH * SLEN * HDIM;     // [B*S,128] f16
  _Float16* Vt = Kh + (size_t)BATCH * SLEN * HDIM;     // [B,128,S] f16 (transposed)
  _Float16* Wh = Vt + (size_t)BATCH * SLEN * HDIM;     // [3,128,768] f16

  wconv_kernel<<<dim3((3 * WSZ) / 1024), dim3(256), 0, stream>>>(Wq, Wk, Wv, Wh);

  dim3 g1((BATCH * SLEN) / 128, 3);
  qkv_kernel<<<g1, dim3(256), 0, stream>>>(X, pmask, Wh, Qh, Kh, Vt);

  dim3 g2(BATCH * (SLEN / 128));
  flash_attn_kernel<<<g2, dim3(256), 0, stream>>>(Qh, Kh, Vt, out);
}
